// CHGNetEmbedding_5085241279007
// MI455X (gfx1250) — matrix operand order repaired and checked by reading, not compiled, not run on hardware
//
#include <hip/hip_runtime.h>
#include <hip/hip_bf16.h>

// ---------------------------------------------------------------------------
// CHGNet-style embedding for MI455X (gfx1250, wave32).
//  - Gated MLPs on v_wmma_f32_16x16x32_bf16 (f32 accumulate).
//  - All big intermediates stored natively bf16 (they are only consumed as
//    bf16 WMMA inputs / epilogue multipliers) -> halves gather bandwidth.
//  - LDS staging of gathered rows uses global_load_async_to_lds_b128
//    (ASYNCcnt) so bytes go memory->LDS without a VGPR round-trip.
//  - Segment sums via hardware fp32 global atomics into f32 accumulators.
// ---------------------------------------------------------------------------

static constexpr int N_  = 50000;
static constexpr int E_  = 400000;
static constexpr int EA_ = 120000;
static constexpr int A_  = 240000;

typedef __attribute__((ext_vector_type(16))) __bf16 v16bf;
typedef __attribute__((ext_vector_type(8)))  float  v8f;
typedef unsigned short ushort_t;

union Frag { v16bf v; uint4 q[2]; };

__device__ __forceinline__ ushort_t f2bf(float x) {
  unsigned u = __float_as_uint(x);
  u += 0x7FFFu + ((u >> 16) & 1u);   // round-to-nearest-even
  return (ushort_t)(u >> 16);
}
__device__ __forceinline__ float b2f(ushort_t h) {
  return __uint_as_float(((unsigned)h) << 16);
}
__device__ __forceinline__ float sigm(float x) { return 1.0f / (1.0f + __expf(-x)); }
__device__ __forceinline__ float silu(float x) { return x * sigm(x); }
__device__ __forceinline__ void atomAddF(float* p, float v) {
  (void)__hip_atomic_fetch_add(p, v, __ATOMIC_RELAXED, __HIP_MEMORY_SCOPE_AGENT);
}
// Async 16B copy global -> LDS (per-lane addresses), tracked by ASYNCcnt.
__device__ __forceinline__ void async_cp16(unsigned lds_off, const void* gptr) {
  asm volatile("global_load_async_to_lds_b128 %0, %1, off"
               :: "v"(lds_off), "v"(gptr) : "memory");
}
__device__ __forceinline__ void async_wait0() {
  asm volatile("s_wait_asynccnt 0" ::: "memory");
}

// --------------------------- weight prep: W(KxN f32) -> W^T(NxK bf16) ------
__global__ void k_wprep(const float* __restrict__ w, ushort_t* __restrict__ wT, int K) {
  int i = blockIdx.x * 256 + threadIdx.x;
  if (i >= 64 * K) return;
  int n = i / K, k = i % K;
  wT[i] = f2bf(w[(size_t)k * 64 + n]);
}

// --------------------------- atom embedding: vi = zi @ vi0_w + b -----------
__global__ void k_embed(const int* __restrict__ species, const float* __restrict__ table,
                        const float* __restrict__ w, const float* __restrict__ b,
                        float* __restrict__ vi, ushort_t* __restrict__ viB) {
  int i = blockIdx.x * 256 + threadIdx.x;   // N*64 threads
  int atom = i >> 6, c = i & 63;
  const float* z = table + (size_t)species[atom] * 16;
  float acc = b[c];
#pragma unroll
  for (int k = 0; k < 16; ++k) acc += z[k] * w[k * 64 + c];
  vi[i]  = acc;
  viB[i] = f2bf(acc);
}

// --------------------------- radial basis -> eijB / eijaB (bf16) -----------
__global__ __launch_bounds__(256)
void k_radial(const float* __restrict__ dist, const float* __restrict__ sw,
              const float* __restrict__ w8x128, ushort_t* __restrict__ eijB,
              ushort_t* __restrict__ eijaB) {
  __shared__ float srb[128 * 8];
  __shared__ float swt[8 * 128];
  int t = threadIdx.x;
  int base = blockIdx.x * 128;
  for (int i = t; i < 1024; i += 256) swt[i] = w8x128[i];
  if (t < 128) {
    int e = base + t;
    float d = dist[e];
    float s = sw[e] * 0.6324555320336759f / d;        // sqrt(2/5)/d * switch
#pragma unroll
    for (int j = 0; j < 8; ++j)
      srb[t * 8 + j] = __sinf((float)(j + 1) * 0.6283185307179586f * d) * s;  // pi/5
  }
  __syncthreads();
  for (int i = t; i < 128 * 128; i += 256) {
    int r = i >> 7, c = i & 127;
    float acc = 0.f;
#pragma unroll
    for (int j = 0; j < 8; ++j) acc += srb[r * 8 + j] * swt[j * 128 + c];
    int e = base + r;
    if (c < 64) eijB[(size_t)e * 64 + c]         = f2bf(acc);
    else        eijaB[(size_t)e * 64 + (c - 64)] = f2bf(acc);
  }
}

// --------------------------- angular-edge radial basis -> eijbB ------------
__global__ __launch_bounds__(256)
void k_radang(const float* __restrict__ dist, const int* __restrict__ filt,
              const float* __restrict__ swa, const float* __restrict__ w8x64,
              ushort_t* __restrict__ eijbB) {
  __shared__ float srb[128 * 8];
  __shared__ float swt[8 * 64];
  int t = threadIdx.x;
  int base = blockIdx.x * 128;
  for (int i = t; i < 512; i += 256) swt[i] = w8x64[i];
  if (t < 128) {
    int i = base + t;
    if (i < EA_) {
      float d = dist[filt[i]];
      float s = swa[i] * 0.6324555320336759f / d;
#pragma unroll
      for (int j = 0; j < 8; ++j)
        srb[t * 8 + j] = __sinf((float)(j + 1) * 0.6283185307179586f * d) * s;
    } else {
#pragma unroll
      for (int j = 0; j < 8; ++j) srb[t * 8 + j] = 0.f;
    }
  }
  __syncthreads();
  for (int i = t; i < 128 * 64; i += 256) {
    int r = i >> 6, c = i & 63;
    int o = base + r;
    if (o >= EA_) continue;
    float acc = 0.f;
#pragma unroll
    for (int j = 0; j < 8; ++j) acc += srb[r * 8 + j] * swt[j * 64 + c];
    eijbB[(size_t)o * 64 + c] = f2bf(acc);
  }
}

// --------------------------- angular basis -> aijkB / aikjB ----------------
__global__ __launch_bounds__(256)
void k_ang(const float* __restrict__ angles, const float* __restrict__ w9x64,
           ushort_t* __restrict__ aijkB, ushort_t* __restrict__ aikjB) {
  __shared__ float sb[64 * 9];
  __shared__ float swt[9 * 64];
  int t = threadIdx.x;
  int base = blockIdx.x * 64;
  for (int i = t; i < 576; i += 256) swt[i] = w9x64[i];
  if (t < 64) {
    float a = angles[base + t];
    sb[t * 9 + 0] = 1.f;
#pragma unroll
    for (int n = 1; n <= 4; ++n) {
      sb[t * 9 + n]     = __cosf((float)n * a);
      sb[t * 9 + 4 + n] = __sinf((float)n * a);
    }
  }
  __syncthreads();
  for (int i = t; i < 64 * 64; i += 256) {
    int r = i >> 6, c = i & 63;
    float acc = 0.f;
#pragma unroll
    for (int j = 0; j < 9; ++j) acc += sb[r * 9 + j] * swt[j * 64 + c];
    size_t o = (size_t)(base + r) * 64 + c;
    ushort_t h = f2bf(acc);
    aijkB[o] = h;
    aikjB[o] = h;
  }
}

// --------------------------- eijkbB = eijbB[src] * eijbB[dst] --------------
__global__ void k_eijkb(const ushort_t* __restrict__ eijbB, const int* __restrict__ asrc,
                        const int* __restrict__ adst, ushort_t* __restrict__ out) {
  int i = blockIdx.x * 256 + threadIdx.x;   // A*64 threads
  int a = i >> 6, c = i & 63;
  out[i] = f2bf(b2f(eijbB[(size_t)asrc[a] * 64 + c]) * b2f(eijbB[(size_t)adst[a] * 64 + c]));
}

// --------------------------- eij_angB = eijB[filter] -----------------------
__global__ void k_gatherEA(const ushort_t* __restrict__ eijB, const int* __restrict__ filt,
                           ushort_t* __restrict__ out) {
  int i = blockIdx.x * 256 + threadIdx.x;   // EA*64 threads
  out[i] = eijB[(size_t)filt[i >> 6] * 64 + (i & 63)];
}

// ---------------------------------------------------------------------------
// phiv: gated MLP over edges, K=192 = [viB[src] | viB[dst] | eijB], M-tile 128.
// Rows staged to LDS with per-lane global_load_async_to_lds_b128.
// out = sigmoid(x@wg+bg)*silu(x@wv+bv)*eija ; scatter-add to agg[edge_src].
// ---------------------------------------------------------------------------
__global__ __launch_bounds__(256)
void k_phiv(const ushort_t* __restrict__ viB, const int* __restrict__ esrc,
            const int* __restrict__ edst, const ushort_t* __restrict__ eijB,
            const ushort_t* __restrict__ eijaB,
            const ushort_t* __restrict__ wgT, const ushort_t* __restrict__ wvT,
            const float* __restrict__ bg, const float* __restrict__ bv,
            float* __restrict__ agg) {
  constexpr int K = 192, KP = 200, M = 128;
  __shared__ ushort_t sx[M * KP];     // 50 KB; row stride 400B (16B aligned)
  int base = blockIdx.x * M;
  int t = threadIdx.x;

  // ---- async stage: thread t -> row (t&127), half (t>>7) -> 12 x 16B chunks
  {
    int row = t & 127, half = t >> 7;
    int e = base + row;
    int es = esrc[e], ed = edst[e];
    unsigned lb = (unsigned)(uintptr_t)sx + (unsigned)(row * KP * 2);
#pragma unroll
    for (int j = 0; j < 12; ++j) {
      int k16 = half * 12 + j;            // 16B chunk index within row (0..23)
      const ushort_t* g;
      if (k16 < 8)       g = viB  + (size_t)es * 64 + k16 * 8;
      else if (k16 < 16) g = viB  + (size_t)ed * 64 + (k16 - 8) * 8;
      else               g = eijB + (size_t)e  * 64 + (k16 - 16) * 8;
      async_cp16(lb + k16 * 16, g);
    }
  }
  async_wait0();
  __syncthreads();

  int lane = t & 31, w = t >> 5;
  int hi = lane >> 4, ln = lane & 15;
  __builtin_prefetch(wgT, 0, 3);
  v8f accg[4] = {}; v8f accv[4] = {};
  int rowA = w * 16 + ln;
#pragma unroll
  for (int k0 = 0; k0 < K; k0 += 32) {
    Frag a;
    int o = k0 + hi * 8;
    a.q[0] = *reinterpret_cast<const uint4*>(&sx[rowA * KP + o]);
    a.q[1] = *reinterpret_cast<const uint4*>(&sx[rowA * KP + o + 16]);
    int ob = k0 + hi * 8;
#pragma unroll
    for (int nt = 0; nt < 4; ++nt) {
      int n = nt * 16 + ln;
      Frag bG, bV;
      const uint4* pg = reinterpret_cast<const uint4*>(wgT + (size_t)n * K + ob);
      const uint4* pv = reinterpret_cast<const uint4*>(wvT + (size_t)n * K + ob);
      bG.q[0] = pg[0]; bG.q[1] = pg[2];
      bV.q[0] = pv[0]; bV.q[1] = pv[2];
      accg[nt] = __builtin_amdgcn_wmma_f32_16x16x32_bf16(false, a.v, false, bG.v,
                                                         (short)0, accg[nt], false, false);
      accv[nt] = __builtin_amdgcn_wmma_f32_16x16x32_bf16(false, a.v, false, bV.v,
                                                         (short)0, accv[nt], false, false);
    }
  }

  int es8[8];
#pragma unroll
  for (int r = 0; r < 8; ++r) es8[r] = esrc[base + w * 16 + hi * 8 + r];
#pragma unroll
  for (int nt = 0; nt < 4; ++nt) {
    int col = nt * 16 + ln;
    float bgc = bg[col], bvc = bv[col];
#pragma unroll
    for (int r = 0; r < 8; ++r) {
      int ee = base + w * 16 + hi * 8 + r;
      float g  = sigm(accg[nt][r] + bgc);
      float vv = silu(accv[nt][r] + bvc);
      float ov = g * vv * b2f(eijaB[(size_t)ee * 64 + col]);
      atomAddF(&agg[(size_t)es8[r] * 64 + col], ov);
    }
  }
}

// ---------------------------------------------------------------------------
// Angle gated MLP, K=256 = [eAB[idx0] | eAB[idx1] | aB | viB[central]], M=64.
// scaleB!=null: val *= eijkb[a], scatter-add f32 to updOut[scat[a]]  (phie)
// scaleB==null: aOut[a] += val (bf16 in-place, aOut==aB)             (phia)
// ---------------------------------------------------------------------------
__global__ __launch_bounds__(256)
void k_gangle(const ushort_t* __restrict__ eAB, const int* __restrict__ idx0,
              const int* __restrict__ idx1, const ushort_t* aB,
              const ushort_t* __restrict__ viB, const int* __restrict__ central,
              const ushort_t* __restrict__ wgT, const ushort_t* __restrict__ wvT,
              const float* __restrict__ bg, const float* __restrict__ bv,
              const ushort_t* __restrict__ scaleB, const int* __restrict__ scat,
              float* updOut, ushort_t* aOut) {
  constexpr int K = 256, KP = 264, M = 64;
  __shared__ ushort_t sx[M * KP];     // 33.8 KB; row stride 528B (16B aligned)
  int base = blockIdx.x * M;
  int t = threadIdx.x;

  // ---- async stage: thread t -> row (t&63), segment (t>>6) -> 8 x 16B chunks
  {
    int row = t & 63, q = t >> 6;
    int a = base + row;
    const ushort_t* g;
    if      (q == 0) g = eAB + (size_t)idx0[a] * 64;
    else if (q == 1) g = eAB + (size_t)idx1[a] * 64;
    else if (q == 2) g = aB  + (size_t)a * 64;
    else             g = viB + (size_t)central[a] * 64;
    unsigned lb = (unsigned)(uintptr_t)sx + (unsigned)((row * KP + q * 64) * 2);
#pragma unroll
    for (int j = 0; j < 8; ++j) async_cp16(lb + j * 16, g + j * 8);
  }
  async_wait0();
  __syncthreads();

  int lane = t & 31, w = t >> 5;
  int hi = lane >> 4, ln = lane & 15;
  int mt = w >> 1;                 // 4 M-tiles, 2 waves each
  int ntb = (w & 1) * 2;           // N-tiles {0,1} or {2,3}
  v8f accg[2] = {}; v8f accv[2] = {};
  int rowA = mt * 16 + ln;
#pragma unroll
  for (int k0 = 0; k0 < K; k0 += 32) {
    Frag a;
    int o = k0 + hi * 8;
    a.q[0] = *reinterpret_cast<const uint4*>(&sx[rowA * KP + o]);
    a.q[1] = *reinterpret_cast<const uint4*>(&sx[rowA * KP + o + 16]);
    int ob = k0 + hi * 8;
#pragma unroll
    for (int j = 0; j < 2; ++j) {
      int n = (ntb + j) * 16 + ln;
      Frag bG, bV;
      const uint4* pg = reinterpret_cast<const uint4*>(wgT + (size_t)n * K + ob);
      const uint4* pv = reinterpret_cast<const uint4*>(wvT + (size_t)n * K + ob);
      bG.q[0] = pg[0]; bG.q[1] = pg[2];
      bV.q[0] = pv[0]; bV.q[1] = pv[2];
      accg[j] = __builtin_amdgcn_wmma_f32_16x16x32_bf16(false, a.v, false, bG.v,
                                                        (short)0, accg[j], false, false);
      accv[j] = __builtin_amdgcn_wmma_f32_16x16x32_bf16(false, a.v, false, bV.v,
                                                        (short)0, accv[j], false, false);
    }
  }

#pragma unroll
  for (int j = 0; j < 2; ++j) {
    int col = (ntb + j) * 16 + ln;
    float bgc = bg[col], bvc = bv[col];
#pragma unroll
    for (int r = 0; r < 8; ++r) {
      int aa = base + mt * 16 + hi * 8 + r;
      float g  = sigm(accg[j][r] + bgc);
      float vv = silu(accv[j][r] + bvc);
      float val = g * vv;
      if (scaleB) {
        val *= b2f(scaleB[(size_t)aa * 64 + col]);
        atomAddF(&updOut[(size_t)scat[aa] * 64 + col], val);
      } else {
        size_t o = (size_t)aa * 64 + col;
        aOut[o] = f2bf(b2f(aOut[o]) + val);
      }
    }
  }
}

// ---------------------------------------------------------------------------
// 64x64 GEMM tail: vi += agg @ vi_w + vi_b ; refresh bf16 mirror viB.
// ---------------------------------------------------------------------------
__global__ __launch_bounds__(256)
void k_viupd(const float* __restrict__ agg, const ushort_t* __restrict__ wT,
             const float* __restrict__ bias, float* __restrict__ vi,
             ushort_t* __restrict__ viB, int Rows) {
  constexpr int K = 64, KP = 72, M = 128;
  __shared__ ushort_t sx[M * KP];     // 18 KB
  int base = blockIdx.x * M;
  int t = threadIdx.x;
  {
    int row = t & 127, half = t >> 7;
    int g = base + row;
#pragma unroll
    for (int j = 0; j < 8; ++j) {
      int c = half * 32 + j * 4;
      float4 f = (g < Rows) ? *reinterpret_cast<const float4*>(agg + (size_t)g * 64 + c)
                            : make_float4(0.f, 0.f, 0.f, 0.f);
      sx[row * KP + c + 0] = f2bf(f.x);
      sx[row * KP + c + 1] = f2bf(f.y);
      sx[row * KP + c + 2] = f2bf(f.z);
      sx[row * KP + c + 3] = f2bf(f.w);
    }
  }
  __syncthreads();

  int lane = t & 31, w = t >> 5;
  int hi = lane >> 4, ln = lane & 15;
  v8f acc[4] = {};
  int rowA = w * 16 + ln;
#pragma unroll
  for (int k0 = 0; k0 < K; k0 += 32) {
    Frag a;
    int o = k0 + hi * 8;
    a.q[0] = *reinterpret_cast<const uint4*>(&sx[rowA * KP + o]);
    a.q[1] = *reinterpret_cast<const uint4*>(&sx[rowA * KP + o + 16]);
    int ob = k0 + hi * 8;
#pragma unroll
    for (int nt = 0; nt < 4; ++nt) {
      Frag b;
      const uint4* pb = reinterpret_cast<const uint4*>(wT + (size_t)(nt * 16 + ln) * K + ob);
      b.q[0] = pb[0]; b.q[1] = pb[2];
      acc[nt] = __builtin_amdgcn_wmma_f32_16x16x32_bf16(false, a.v, false, b.v,
                                                        (short)0, acc[nt], false, false);
    }
  }
#pragma unroll
  for (int nt = 0; nt < 4; ++nt) {
    int col = nt * 16 + ln;
    float bc = bias[col];
#pragma unroll
    for (int r = 0; r < 8; ++r) {
      int rr = base + w * 16 + hi * 8 + r;
      if (rr < Rows) {
        size_t o = (size_t)rr * 64 + col;
        float nv = vi[o] + acc[nt][r] + bc;
        vi[o]  = nv;
        viB[o] = f2bf(nv);
      }
    }
  }
}

// ---------------------------------------------------------------------------
// 64x64 GEMM tail: eij_ang += upd @ le_w ; write back bf16 eij[filter].
// ---------------------------------------------------------------------------
__global__ __launch_bounds__(256)
void k_leupd(const float* __restrict__ upd, const ushort_t* __restrict__ wT,
             ushort_t* __restrict__ eijangB, ushort_t* __restrict__ eijB,
             const int* __restrict__ filt, int Rows) {
  constexpr int K = 64, KP = 72, M = 128;
  __shared__ ushort_t sx[M * KP];
  int base = blockIdx.x * M;
  int t = threadIdx.x;
  {
    int row = t & 127, half = t >> 7;
    int g = base + row;
#pragma unroll
    for (int j = 0; j < 8; ++j) {
      int c = half * 32 + j * 4;
      float4 f = (g < Rows) ? *reinterpret_cast<const float4*>(upd + (size_t)g * 64 + c)
                            : make_float4(0.f, 0.f, 0.f, 0.f);
      sx[row * KP + c + 0] = f2bf(f.x);
      sx[row * KP + c + 1] = f2bf(f.y);
      sx[row * KP + c + 2] = f2bf(f.z);
      sx[row * KP + c + 3] = f2bf(f.w);
    }
  }
  __syncthreads();

  int lane = t & 31, w = t >> 5;
  int hi = lane >> 4, ln = lane & 15;
  v8f acc[4] = {};
  int rowA = w * 16 + ln;
#pragma unroll
  for (int k0 = 0; k0 < K; k0 += 32) {
    Frag a;
    int o = k0 + hi * 8;
    a.q[0] = *reinterpret_cast<const uint4*>(&sx[rowA * KP + o]);
    a.q[1] = *reinterpret_cast<const uint4*>(&sx[rowA * KP + o + 16]);
    int ob = k0 + hi * 8;
#pragma unroll
    for (int nt = 0; nt < 4; ++nt) {
      Frag b;
      const uint4* pb = reinterpret_cast<const uint4*>(wT + (size_t)(nt * 16 + ln) * K + ob);
      b.q[0] = pb[0]; b.q[1] = pb[2];
      acc[nt] = __builtin_amdgcn_wmma_f32_16x16x32_bf16(false, a.v, false, b.v,
                                                        (short)0, acc[nt], false, false);
    }
  }
#pragma unroll
  for (int nt = 0; nt < 4; ++nt) {
    int col = nt * 16 + ln;
#pragma unroll
    for (int r = 0; r < 8; ++r) {
      int rr = base + w * 16 + hi * 8 + r;
      if (rr < Rows) {
        size_t o = (size_t)rr * 64 + col;
        float v = b2f(eijangB[o]) + acc[nt][r];
        ushort_t h = f2bf(v);
        eijangB[o] = h;
        eijB[(size_t)filt[rr] * 64 + col] = h;
      }
    }
  }
}

// ---------------------------------------------------------------------------
// Host orchestration. Input layout: params (jax tree-flatten, alphabetical
// dict keys) then remaining setup_inputs entries in insertion order:
//  0 aijk0_w | 1 eij0_w | 2 eijb_w
//  layer l in {0,1} at base 3+15l: le_w, phia.{bg,bv,wg,wv}, phie.{bg,bv,wg,wv},
//                                  phiv.{bg,bv,wg,wv}, vi_b, vi_w
//  layer2 at 33: phiv.{bg,bv,wg,wv}, vi_b, vi_w
//  39 species_table | 40 vi0_b | 41 vi0_w
//  42 species 43 edge_src 44 edge_dst 45 distances 46 switch 47 angles
//  48 angle_src 49 angle_dst 50 switch_angles 51 central_atom 52 filter_indices
// ---------------------------------------------------------------------------
extern "C" void kernel_launch(void* const* d_in, const int* in_sizes, int n_in,
                              void* d_out, int out_size, void* d_ws, size_t ws_size,
                              hipStream_t stream) {
  (void)in_sizes; (void)n_in; (void)out_size; (void)ws_size;
  auto F = [&](int i) { return (const float*)d_in[i]; };
  auto I = [&](int i) { return (const int*)d_in[i]; };

  const float* w_aijk0 = F(0);
  const float* w_eij0  = F(1);
  const float* w_eijb  = F(2);
  const float* phiv_wg[3] = {F(14), F(29), F(35)};
  const float* phiv_wv[3] = {F(15), F(30), F(36)};
  const float* phiv_bg[3] = {F(12), F(27), F(33)};
  const float* phiv_bv[3] = {F(13), F(28), F(34)};
  const float* vi_w[3]    = {F(17), F(32), F(38)};
  const float* vi_b[3]    = {F(16), F(31), F(37)};
  const float* phie_wg[2] = {F(10), F(25)};
  const float* phie_wv[2] = {F(11), F(26)};
  const float* phie_bg[2] = {F(8),  F(23)};
  const float* phie_bv[2] = {F(9),  F(24)};
  const float* phia_wg[2] = {F(6),  F(21)};
  const float* phia_wv[2] = {F(7),  F(22)};
  const float* phia_bg[2] = {F(4),  F(19)};
  const float* phia_bv[2] = {F(5),  F(20)};
  const float* le_w[2]    = {F(3),  F(18)};
  const float* table = F(39);
  const float* vi0_b = F(40);
  const float* vi0_w = F(41);
  const int*   species  = I(42);
  const int*   edge_src = I(43);
  const int*   edge_dst = I(44);
  const float* dist     = F(45);
  const float* sw       = F(46);
  const float* angles   = F(47);
  const int*   ang_src  = I(48);
  const int*   ang_dst  = I(49);
  const float* sw_ang   = F(50);
  const int*   central  = I(51);
  const int*   filt     = I(52);

  float* vi = (float*)d_out;

  // workspace bump allocator
  uintptr_t cur = (uintptr_t)d_ws;
  auto take = [&](size_t b) -> void* {
    void* r = (void*)cur;
    cur += (b + 255) & ~(size_t)255;
    return r;
  };
  // bf16 masters/mirrors
  ushort_t* eijB    = (ushort_t*)take((size_t)E_  * 64 * 2);
  ushort_t* eijaB   = (ushort_t*)take((size_t)E_  * 64 * 2);
  ushort_t* eijbB   = (ushort_t*)take((size_t)EA_ * 64 * 2);
  ushort_t* eijkbB  = (ushort_t*)take((size_t)A_  * 64 * 2);
  ushort_t* aijkB   = (ushort_t*)take((size_t)A_  * 64 * 2);
  ushort_t* aikjB   = (ushort_t*)take((size_t)A_  * 64 * 2);
  ushort_t* eijangB = (ushort_t*)take((size_t)EA_ * 64 * 2);
  ushort_t* viB     = (ushort_t*)take((size_t)N_  * 64 * 2);
  // f32 accumulators
  float* agg = (float*)take((size_t)N_  * 64 * 4);
  float* upd = (float*)take((size_t)EA_ * 64 * 4);
  // bf16 transposed weights
  ushort_t *bphiv_g[3], *bphiv_v[3], *bviw[3];
  ushort_t *bphie_g[2], *bphie_v[2], *bphia_g[2], *bphia_v[2], *blew[2];
  for (int l = 0; l < 3; ++l) {
    bphiv_g[l] = (ushort_t*)take(64 * 192 * 2);
    bphiv_v[l] = (ushort_t*)take(64 * 192 * 2);
    bviw[l]    = (ushort_t*)take(64 * 64 * 2);
  }
  for (int l = 0; l < 2; ++l) {
    bphie_g[l] = (ushort_t*)take(64 * 256 * 2);
    bphie_v[l] = (ushort_t*)take(64 * 256 * 2);
    bphia_g[l] = (ushort_t*)take(64 * 256 * 2);
    bphia_v[l] = (ushort_t*)take(64 * 256 * 2);
    blew[l]    = (ushort_t*)take(64 * 64 * 2);
  }

  auto prep = [&](const float* w, ushort_t* wT, int K) {
    k_wprep<<<(64 * K + 255) / 256, 256, 0, stream>>>(w, wT, K);
  };
  for (int l = 0; l < 3; ++l) {
    prep(phiv_wg[l], bphiv_g[l], 192);
    prep(phiv_wv[l], bphiv_v[l], 192);
    prep(vi_w[l], bviw[l], 64);
  }
  for (int l = 0; l < 2; ++l) {
    prep(phie_wg[l], bphie_g[l], 256);
    prep(phie_wv[l], bphie_v[l], 256);
    prep(phia_wg[l], bphia_g[l], 256);
    prep(phia_wv[l], bphia_v[l], 256);
    prep(le_w[l], blew[l], 64);
  }

  // embeddings / bases
  k_embed <<<N_ * 64 / 256, 256, 0, stream>>>(species, table, vi0_w, vi0_b, vi, viB);
  k_radial<<<E_ / 128,      256, 0, stream>>>(dist, sw, w_eij0, eijB, eijaB);
  k_radang<<<(EA_ + 127) / 128, 256, 0, stream>>>(dist, filt, sw_ang, w_eijb, eijbB);
  k_eijkb <<<A_ * 64 / 256, 256, 0, stream>>>(eijbB, ang_src, ang_dst, eijkbB);
  k_ang   <<<A_ / 64,       256, 0, stream>>>(angles, w_aijk0, aijkB, aikjB);

  for (int l = 0; l < 3; ++l) {
    hipMemsetAsync(agg, 0, (size_t)N_ * 64 * 4, stream);
    k_phiv<<<E_ / 128, 256, 0, stream>>>(viB, edge_src, edge_dst, eijB, eijaB,
                                         bphiv_g[l], bphiv_v[l], phiv_bg[l], phiv_bv[l], agg);
    k_viupd<<<(N_ + 127) / 128, 256, 0, stream>>>(agg, bviw[l], vi_b[l], vi, viB, N_);
    if (l == 2) break;

    k_gatherEA<<<EA_ * 64 / 256, 256, 0, stream>>>(eijB, filt, eijangB);
    hipMemsetAsync(upd, 0, (size_t)EA_ * 64 * 4, stream);
    // phie (ijk): x=[es,ed,aijk,vi_c], scatter to angle_src
    k_gangle<<<A_ / 64, 256, 0, stream>>>(eijangB, ang_src, ang_dst, aijkB, viB, central,
                                          bphie_g[l], bphie_v[l], phie_bg[l], phie_bv[l],
                                          eijkbB, ang_src, upd, nullptr);
    // phie (ikj): x=[ed,es,aikj,vi_c], scatter to angle_dst
    k_gangle<<<A_ / 64, 256, 0, stream>>>(eijangB, ang_dst, ang_src, aikjB, viB, central,
                                          bphie_g[l], bphie_v[l], phie_bg[l], phie_bv[l],
                                          eijkbB, ang_dst, upd, nullptr);
    k_leupd<<<(EA_ + 127) / 128, 256, 0, stream>>>(upd, blew[l], eijangB, eijB, filt, EA_);
    // phia: aijk/aikj += gated(...), using updated eij_ang
    k_gangle<<<A_ / 64, 256, 0, stream>>>(eijangB, ang_src, ang_dst, aijkB, viB, central,
                                          bphia_g[l], bphia_v[l], phia_bg[l], phia_bv[l],
                                          nullptr, nullptr, nullptr, aijkB);
    k_gangle<<<A_ / 64, 256, 0, stream>>>(eijangB, ang_dst, ang_src, aikjB, viB, central,
                                          bphia_g[l], bphia_v[l], phia_bg[l], phia_bv[l],
                                          nullptr, nullptr, nullptr, aikjB);
  }
}
